// MultiHeadAttention_2336462209747
// MI455X (gfx1250) — matrix operand order repaired and checked by reading, not compiled, not run on hardware
//
#include <hip/hip_runtime.h>
#include <hip/hip_bf16.h>
#include <math.h>

// ---------- CDNA5 WMMA types ----------
typedef __attribute__((ext_vector_type(16))) _Float16 v16h;
typedef __attribute__((ext_vector_type(8)))  float    v8f;

union FragH {           // one 16x32 (A) or 32x16 (B) f16 fragment: 8 VGPRs
  uint4 u[2];
  v16h  v;
};
union H8 { _Float16 h[8]; uint4 u; };

// ---------- problem constants ----------
constexpr int Bb   = 2;
constexpr int S    = 2048;
constexpr int Hd   = 1024;
constexpr int H    = 16;
constexpr int Dh   = 64;
constexpr int Mtot = Bb * S;   // 4096
constexpr int Ktot = Hd;       // 1024 contraction for projections

// ---------- wave32 16-lane butterfly reductions (stay within half-wave) ----------
__device__ __forceinline__ float redmax16(float v) {
#pragma unroll
  for (int m = 1; m < 16; m <<= 1) v = fmaxf(v, __shfl_xor(v, m, 32));
  return v;
}
__device__ __forceinline__ float redsum16(float v) {
#pragma unroll
  for (int m = 1; m < 16; m <<= 1) v += __shfl_xor(v, m, 32);
  return v;
}

// ---------- f32 -> f16 conversion ----------
__global__ void mha_cvt_f32_f16(const float* __restrict__ src,
                                _Float16* __restrict__ dst, int n) {
  int i  = blockIdx.x * blockDim.x + threadIdx.x;
  int st = gridDim.x * blockDim.x;
  for (; i < n; i += st) dst[i] = (_Float16)src[i];
}

// ---------- GEMM: C = X[4096,1024] @ W^T (W is [1024,1024] row-major) ----------
// Each wave computes a 16x64 output strip (4 accumulators, A-frag reused 4x).
// MODE 0: f16 out, head-split layout [B,H,S,Dh]          (Q, K projections)
// MODE 1: f16 out, transposed head layout [B,H,Dh,S]     (V projection)
// MODE 2: f32 out, row-major [M,N]                       (final Wo GEMM)
template <int MODE>
__global__ __launch_bounds__(256) void mha_gemm_xwT(const _Float16* __restrict__ X,
                                                    const _Float16* __restrict__ W,
                                                    void* __restrict__ OutP) {
  const int wave = threadIdx.x >> 5;
  const int lane = threadIdx.x & 31;
  const int lm   = lane & 15;
  const int lh   = lane >> 4;
  const int w    = blockIdx.x * 8 + wave;   // 4096 waves total
  const int m0   = (w >> 4) * 16;           // 256 m-tiles
  const int n0   = (w & 15) * 64;           // 16 n-strips

  v8f acc[4];
#pragma unroll
  for (int t = 0; t < 4; ++t) { v8f z = {}; acc[t] = z; }

  const _Float16* arow = X + (size_t)(m0 + lm) * Ktot + lh * 16;
  for (int k0 = 0; k0 < Ktot; k0 += 32) {
    FragH fa;   // A: X rows, K = k0..k0+31
    fa.u[0] = *reinterpret_cast<const uint4*>(arow + k0);
    fa.u[1] = *reinterpret_cast<const uint4*>(arow + k0 + 8);
#pragma unroll
    for (int t = 0; t < 4; ++t) {
      const _Float16* bp = W + (size_t)(n0 + t * 16 + lm) * Ktot + k0 + lh * 16;
      FragH fb;  // B[k][n] = W[n][k]: lane = output column n, contiguous K run
      fb.u[0] = *reinterpret_cast<const uint4*>(bp);
      fb.u[1] = *reinterpret_cast<const uint4*>(bp + 8);
      acc[t] = __builtin_amdgcn_wmma_f32_16x16x32_f16(false, fa.v, false, fb.v,
                                                      (short)0, acc[t], false, false);
    }
  }

  if constexpr (MODE == 0) {
    _Float16* O = reinterpret_cast<_Float16*>(OutP);
#pragma unroll
    for (int t = 0; t < 4; ++t) {
      const int n = n0 + t * 16 + lm, hh = n >> 6, dl = n & 63;
#pragma unroll
      for (int r = 0; r < 8; ++r) {
        const int mg = m0 + r + 8 * lh, b = mg >> 11, s = mg & (S - 1);
        O[(size_t)((b * H + hh) * S + s) * Dh + dl] = (_Float16)acc[t][r];
      }
    }
  } else if constexpr (MODE == 1) {
    _Float16* O = reinterpret_cast<_Float16*>(OutP);
    const int b = m0 >> 11, sbase = (m0 & (S - 1)) + 8 * lh;
#pragma unroll
    for (int t = 0; t < 4; ++t) {
      const int n = n0 + t * 16 + lm, hh = n >> 6, dl = n & 63;
      H8 tmp;
#pragma unroll
      for (int r = 0; r < 8; ++r) tmp.h[r] = (_Float16)acc[t][r];
      *reinterpret_cast<uint4*>(O + (size_t)((b * H + hh) * Dh + dl) * S + sbase) = tmp.u;
    }
  } else {
    float* O = reinterpret_cast<float*>(OutP);
#pragma unroll
    for (int t = 0; t < 4; ++t)
#pragma unroll
      for (int r = 0; r < 8; ++r)
        O[(size_t)(m0 + r + 8 * lh) * Hd + n0 + t * 16 + lm] = acc[t][r];
  }
}

// ---------- flash attention: one 16-row q-tile per wave, k in blocks of 32 ----------
__global__ __launch_bounds__(256) void mha_flash_attn(const _Float16* __restrict__ Qh,
                                                      const _Float16* __restrict__ Kh,
                                                      const _Float16* __restrict__ Vt,
                                                      const unsigned char* __restrict__ mask,
                                                      _Float16* __restrict__ Oh) {
  __shared__ alignas(16) _Float16 plds[8][16 * 32];   // 1 KB P-tile per wave

  const int wave = threadIdx.x >> 5;
  const int lane = threadIdx.x & 31;
  const int lm   = lane & 15;
  const int lh   = lane >> 4;
  const int w    = blockIdx.x * 8 + wave;             // 4096 q-tiles total
  const int b    = w >> 11;                           // / (H * S/16)
  const int h    = (w >> 7) & 15;
  const int q0   = (w & 127) << 4;

  const _Float16* Qb = Qh + (size_t)((b * H + h) * S) * Dh;
  const _Float16* Kb = Kh + (size_t)((b * H + h) * S) * Dh;
  const _Float16* Vb = Vt + (size_t)((b * H + h) * Dh) * S;
  const unsigned char* Mb = mask + (size_t)b * S * S;

  // Q A-frags over d = 0..31 and 32..63, loaded once
  FragH aq0, aq1;
  const _Float16* qrow = Qb + (size_t)(q0 + lm) * Dh + lh * 16;
  aq0.u[0] = *reinterpret_cast<const uint4*>(qrow);
  aq0.u[1] = *reinterpret_cast<const uint4*>(qrow + 8);
  aq1.u[0] = *reinterpret_cast<const uint4*>(qrow + 32);
  aq1.u[1] = *reinterpret_cast<const uint4*>(qrow + 40);

  float mrun[8], lrun[8];
  v8f o[4];
#pragma unroll
  for (int r = 0; r < 8; ++r) { mrun[r] = -INFINITY; lrun[r] = 0.f; }
#pragma unroll
  for (int j = 0; j < 4; ++j) { v8f z = {}; o[j] = z; }

  const float scale = 0.125f;   // 1/sqrt(64)

  for (int kb = 0; kb < S; kb += 32) {
    // K B-frags: lane = key column; contiguous d runs thanks to [B,H,S,d] layout
    const _Float16* kr0 = Kb + (size_t)(kb + lm) * Dh + lh * 16;
    const _Float16* kr1 = Kb + (size_t)(kb + 16 + lm) * Dh + lh * 16;
    FragH b00, b01, b10, b11;
    b00.u[0] = *reinterpret_cast<const uint4*>(kr0);
    b00.u[1] = *reinterpret_cast<const uint4*>(kr0 + 8);
    b01.u[0] = *reinterpret_cast<const uint4*>(kr0 + 32);
    b01.u[1] = *reinterpret_cast<const uint4*>(kr0 + 40);
    b10.u[0] = *reinterpret_cast<const uint4*>(kr1);
    b10.u[1] = *reinterpret_cast<const uint4*>(kr1 + 8);
    b11.u[0] = *reinterpret_cast<const uint4*>(kr1 + 32);
    b11.u[1] = *reinterpret_cast<const uint4*>(kr1 + 40);

    v8f t0 = {}, t1 = {};
    t0 = __builtin_amdgcn_wmma_f32_16x16x32_f16(false, aq0.v, false, b00.v, (short)0, t0, false, false);
    t0 = __builtin_amdgcn_wmma_f32_16x16x32_f16(false, aq1.v, false, b01.v, (short)0, t0, false, false);
    t1 = __builtin_amdgcn_wmma_f32_16x16x32_f16(false, aq0.v, false, b10.v, (short)0, t1, false, false);
    t1 = __builtin_amdgcn_wmma_f32_16x16x32_f16(false, aq1.v, false, b11.v, (short)0, t1, false, false);

    if (kb + 32 < S) {  // prefetch next K tile -> global_prefetch_b8
      __builtin_prefetch(Kb + (size_t)(kb + 32 + lm) * Dh, 0, 3);
      __builtin_prefetch(Kb + (size_t)(kb + 48 + lm) * Dh, 0, 3);
    }

    // mask (before scale, as in reference), then online softmax per row
#pragma unroll
    for (int r = 0; r < 8; ++r) {
      const int qr = q0 + r + 8 * lh;
      const unsigned char* mrow = Mb + (size_t)qr * S + kb;
      float s0 = (mrow[lm]      ? -1e9f : t0[r]) * scale;
      float s1 = (mrow[16 + lm] ? -1e9f : t1[r]) * scale;
      const float tm    = redmax16(fmaxf(s0, s1));
      const float mnew  = fmaxf(mrun[r], tm);
      const float alpha = __expf(mrun[r] - mnew);
      mrun[r] = mnew;
      const float p0 = __expf(s0 - mnew);
      const float p1 = __expf(s1 - mnew);
      lrun[r] = lrun[r] * alpha + redsum16(p0 + p1);
      o[0][r] *= alpha; o[1][r] *= alpha; o[2][r] *= alpha; o[3][r] *= alpha;
      const int ql = r + 8 * lh;                 // C-layout -> LDS [q][k] tile
      plds[wave][ql * 32 + lm]      = (_Float16)p0;
      plds[wave][ql * 32 + 16 + lm] = (_Float16)p1;
    }
    __syncthreads();

    // reload P as an A-frag (lane = q row) — the required 16-lane transpose
    FragH pa;
    const _Float16* pl = &plds[wave][lm * 32 + lh * 16];
    pa.u[0] = *reinterpret_cast<const uint4*>(pl);
    pa.u[1] = *reinterpret_cast<const uint4*>(pl + 8);
    __syncthreads();   // WAR fence before next iteration's stores

    // P @ V over 4 d-chunks; Vt layout [B,H,d,S] makes B-frags contiguous
#pragma unroll
    for (int j = 0; j < 4; ++j) {
      const _Float16* vp = Vb + (size_t)(j * 16 + lm) * S + kb + lh * 16;
      FragH vb;
      vb.u[0] = *reinterpret_cast<const uint4*>(vp);
      vb.u[1] = *reinterpret_cast<const uint4*>(vp + 8);
      o[j] = __builtin_amdgcn_wmma_f32_16x16x32_f16(false, pa.v, false, vb.v,
                                                    (short)0, o[j], false, false);
    }
  }

  // normalize and write concat-head layout [B,S,Hd] as f16 for the Wo GEMM
#pragma unroll
  for (int j = 0; j < 4; ++j)
#pragma unroll
    for (int r = 0; r < 8; ++r) {
      const int qr = q0 + r + 8 * lh;
      const float val = o[j][r] / lrun[r];
      Oh[(size_t)(b * S + qr) * Hd + h * 64 + j * 16 + lm] = (_Float16)val;
    }
}

// ---------- host-side orchestration ----------
extern "C" void kernel_launch(void* const* d_in, const int* in_sizes, int n_in,
                              void* d_out, int out_size, void* d_ws, size_t ws_size,
                              hipStream_t stream) {
  const float* q  = reinterpret_cast<const float*>(d_in[0]);
  const float* k  = reinterpret_cast<const float*>(d_in[1]);
  const float* v  = reinterpret_cast<const float*>(d_in[2]);
  const unsigned char* mask = reinterpret_cast<const unsigned char*>(d_in[3]);
  const float* Wq = reinterpret_cast<const float*>(d_in[4]);
  const float* Wk = reinterpret_cast<const float*>(d_in[5]);
  const float* Wv = reinterpret_cast<const float*>(d_in[6]);
  const float* Wo = reinterpret_cast<const float*>(d_in[7]);

  char* ws = reinterpret_cast<char*>(d_ws);
  constexpr size_t MB = 1024ull * 1024ull;
  _Float16* q16  = reinterpret_cast<_Float16*>(ws + 0);       // 8 MB
  _Float16* k16  = reinterpret_cast<_Float16*>(ws + 8 * MB);  // 8 MB
  _Float16* v16  = reinterpret_cast<_Float16*>(ws + 16 * MB); // 8 MB
  _Float16* Wq16 = reinterpret_cast<_Float16*>(ws + 24 * MB); // 2 MB
  _Float16* Wk16 = reinterpret_cast<_Float16*>(ws + 26 * MB);
  _Float16* Wv16 = reinterpret_cast<_Float16*>(ws + 28 * MB);
  _Float16* Wo16 = reinterpret_cast<_Float16*>(ws + 30 * MB);
  _Float16* Qh   = reinterpret_cast<_Float16*>(ws + 32 * MB); // 8 MB [B,H,S,d]
  _Float16* Kh   = reinterpret_cast<_Float16*>(ws + 40 * MB); // 8 MB [B,H,S,d]
  _Float16* Vt   = reinterpret_cast<_Float16*>(ws + 48 * MB); // 8 MB [B,H,d,S]
  _Float16* Oh   = reinterpret_cast<_Float16*>(ws + 0);       // alias q16 (dead by then)

  const int NX = Mtot * Hd;   // 4194304
  const int NW = Hd * Hd;     // 1048576

  mha_cvt_f32_f16<<<512, 256, 0, stream>>>(q,  q16,  NX);
  mha_cvt_f32_f16<<<512, 256, 0, stream>>>(k,  k16,  NX);
  mha_cvt_f32_f16<<<512, 256, 0, stream>>>(v,  v16,  NX);
  mha_cvt_f32_f16<<<256, 256, 0, stream>>>(Wq, Wq16, NW);
  mha_cvt_f32_f16<<<256, 256, 0, stream>>>(Wk, Wk16, NW);
  mha_cvt_f32_f16<<<256, 256, 0, stream>>>(Wv, Wv16, NW);
  mha_cvt_f32_f16<<<256, 256, 0, stream>>>(Wo, Wo16, NW);

  mha_gemm_xwT<0><<<512, 256, 0, stream>>>(q16, Wq16, Qh);  // Q projection
  mha_gemm_xwT<0><<<512, 256, 0, stream>>>(k16, Wk16, Kh);  // K projection
  mha_gemm_xwT<1><<<512, 256, 0, stream>>>(v16, Wv16, Vt);  // V projection (transposed)

  mha_flash_attn<<<512, 256, 0, stream>>>(Qh, Kh, Vt, mask, Oh);

  mha_gemm_xwT<2><<<512, 256, 0, stream>>>(Oh, Wo16, d_out); // output projection -> f32
}
